// H2GNN_Variant_59201829208678
// MI455X (gfx1250) — matrix operand order repaired and checked by reading, not compiled, not run on hardware
//
#include <hip/hip_runtime.h>
#include <hip/hip_bf16.h>

// ---------------------------------------------------------------------------
// Types for CDNA5 WMMA (wave32): 16x16x32 bf16 -> f32 accumulate
// ---------------------------------------------------------------------------
typedef __bf16 bf16_t;
typedef bf16_t v16bf __attribute__((ext_vector_type(16)));
typedef float  v8f   __attribute__((ext_vector_type(8)));

#define BM 64
#define BN 32
#define BK 64
#define LDSA (BK + 4)   // 68 bf16 -> 136B row stride (34 banks): conflict-free
#define LDSB (BK + 4)

// ---------------------------------------------------------------------------
// Tiled GEMM: C[m, cofs+n] = (relu?)(sum_k A[m, aofs+k]*B[k, n] + bias[n])
// fp32 in memory; tiles converted to bf16 in LDS; v_wmma_f32_16x16x32_bf16
// with fp32 accumulation. Block: 256 threads = 8 waves = 4(M) x 2(N) tiles.
// Requires K % 64 == 0 (512/256/768 all qualify); M, Nout arbitrary (Nout>=4).
// Staging is branchless: OOB rows/cols are clamped in-bounds; the garbage
// only reaches output rows/cols that the epilogue never stores.
// ---------------------------------------------------------------------------
__global__ __launch_bounds__(256) void gemm_bf16_wmma(
    const float* __restrict__ A, int ldA, int aofs,
    const float* __restrict__ B, int ldB,
    const float* __restrict__ bias,
    float* __restrict__ C, int ldC, int cofs,
    int M, int Nout, int K, int relu_flag)
{
  __shared__ bf16_t As[BM][LDSA];          // [m][k]
  __shared__ bf16_t Bs[BN][LDSB];          // [n][k] (transposed on stage-in)

  const int tid  = threadIdx.x;
  const int lane = tid & 31;
  const int wid  = tid >> 5;
  const int wm   = wid >> 1;               // 0..3  (M direction)
  const int wn   = wid & 1;                // 0..1  (N direction)
  const int m0   = blockIdx.x * BM;
  const int n0   = blockIdx.y * BN;
  const int Mc   = M - 1;
  const int Nc   = Nout - 4;

  v8f acc = {0.f, 0.f, 0.f, 0.f, 0.f, 0.f, 0.f, 0.f};

  for (int k0 = 0; k0 < K; k0 += BK) {
    // --- Stage A tile (BM x BK): 4x float4 per thread, branchless clamp ---
#pragma unroll
    for (int j = 0; j < (BM * BK) / (256 * 4); ++j) {
      int idx = j * 256 + tid;
      int r   = idx >> 4;                  // 0..63  (16 float4 per row)
      int c4  = (idx & 15) * 4;
      int gm  = min(m0 + r, Mc);
      const float4 v = *(const float4*)(A + (size_t)gm * ldA + aofs + k0 + c4);
      As[r][c4 + 0] = (bf16_t)v.x;
      As[r][c4 + 1] = (bf16_t)v.y;
      As[r][c4 + 2] = (bf16_t)v.z;
      As[r][c4 + 3] = (bf16_t)v.w;
    }
    // --- Stage B tile (BK x BN): 2x float4 per thread, stored [n][k] ---
#pragma unroll
    for (int j = 0; j < (BK * BN) / (256 * 4); ++j) {
      int idx = j * 256 + tid;
      int kk  = idx >> 3;                  // 0..63  (8 float4 per row)
      int n4  = (idx & 7) * 4;
      int gn  = min(n0 + n4, Nc);
      const float4 v = *(const float4*)(B + (size_t)(k0 + kk) * ldB + gn);
      Bs[n4 + 0][kk] = (bf16_t)v.x;
      Bs[n4 + 1][kk] = (bf16_t)v.y;
      Bs[n4 + 2][kk] = (bf16_t)v.z;
      Bs[n4 + 3][kk] = (bf16_t)v.w;
    }
    __syncthreads();

    // --- 2 WMMA sub-steps per staged chunk ---
    const int hh = lane >> 4;
    const int la = lane & 15;
#pragma unroll
    for (int ks = 0; ks < BK; ks += 32) {
      // A 16x32 bf16 layout: lane half selects K-octet within each 16-K group
      // B 32x16 bf16 layout: same K order as A (lane half selects K-octet)
      v16bf afrag, bfrag;
#pragma unroll
      for (int i = 0; i < 16; ++i) {
        const int vg = i >> 1;
        const int kA = ((vg >= 4) ? 16 : 0) + hh * 8 + (vg & 3) * 2 + (i & 1);
        afrag[i] = As[wm * 16 + la][ks + kA];
        const int kB = ((i >= 8) ? 16 : 0) + hh * 8 + (i & 7);
        bfrag[i] = Bs[wn * 16 + la][ks + kB];
      }
      acc = __builtin_amdgcn_wmma_f32_16x16x32_bf16(
          /*neg_a=*/false, afrag, /*neg_b=*/false, bfrag,
          /*c_mod=*/(short)0, acc, /*reuse_a=*/false, /*reuse_b=*/false);
    }
    __syncthreads();
  }

  // --- Epilogue: C/D layout -> lane holds column n=lane&15, VGPR r -> row ---
  const int n     = n0 + wn * 16 + (lane & 15);
  const int mrow0 = m0 + wm * 16 + (lane >> 4) * 8;
  if (n < Nout) {
    const float bv = bias ? bias[n] : 0.f;
#pragma unroll
    for (int r = 0; r < 8; ++r) {
      const int m = mrow0 + r;
      if (m < M) {
        float v = acc[r] + bv;
        if (relu_flag) v = fmaxf(v, 0.f);
        C[(size_t)m * ldC + cofs + n] = v;
      }
    }
  }
}

// ---------------------------------------------------------------------------
// Utility kernels
// ---------------------------------------------------------------------------
__global__ void zero_kernel(float* p, long n) {
  long i = (long)blockIdx.x * blockDim.x + threadIdx.x;
  if (i < n) p[i] = 0.f;
}

__global__ void deg_kernel(const int* __restrict__ col, int E, float* deg) {
  int e = blockIdx.x * blockDim.x + threadIdx.x;
  if (e < E)
    __hip_atomic_fetch_add(&deg[col[e]], 1.0f, __ATOMIC_RELAXED,
                           __HIP_MEMORY_SCOPE_AGENT);
}

__global__ void rsqrt_inplace_kernel(float* d, int n) {
  int i = blockIdx.x * blockDim.x + threadIdx.x;
  if (i < n) {
    float v = d[i];
    d[i] = (v > 0.f) ? rsqrtf(v) : 0.f;
  }
}

// Broadcast bias into ncols columns of dst (bias repeats every nb columns).
__global__ void set_cols_kernel(float* __restrict__ dst, int ld,
                                const float* __restrict__ b, int nb,
                                int ncols, int n) {
  long gid = (long)blockIdx.x * blockDim.x + threadIdx.x;
  long total = (long)n * ncols;
  if (gid < total) {
    int i = (int)(gid / ncols);
    int j = (int)(gid % ncols);
    dst[(size_t)i * ld + j] = b[j % nb];
  }
}

// One wave (32 lanes) per edge; 128 features = 32 lanes x float4.
// out points at the destination column-slice base; row stride ldC.
__global__ void gcn_scatter_kernel(const float* __restrict__ hW,
                                   const float* __restrict__ dis,
                                   const int* __restrict__ row,
                                   const int* __restrict__ col,
                                   int E, float* __restrict__ out, int ldC) {
  long gid = (long)blockIdx.x * blockDim.x + threadIdx.x;
  int e    = (int)(gid >> 5);
  int lane = (int)(gid & 31);
  if (e >= E) return;
  const int r = row[e];
  const int c = col[e];
  const float norm = dis[r] * dis[c];
  const float4 v = ((const float4*)(hW + (size_t)r * 128))[lane];
  float* o = out + (size_t)c * ldC + lane * 4;
  __hip_atomic_fetch_add(o + 0, v.x * norm, __ATOMIC_RELAXED, __HIP_MEMORY_SCOPE_AGENT);
  __hip_atomic_fetch_add(o + 1, v.y * norm, __ATOMIC_RELAXED, __HIP_MEMORY_SCOPE_AGENT);
  __hip_atomic_fetch_add(o + 2, v.z * norm, __ATOMIC_RELAXED, __HIP_MEMORY_SCOPE_AGENT);
  __hip_atomic_fetch_add(o + 3, v.w * norm, __ATOMIC_RELAXED, __HIP_MEMORY_SCOPE_AGENT);
}

// In-place log_softmax over rows of width nc (nc=40): one thread per row.
__global__ void log_softmax_kernel(float* __restrict__ out, int n, int nc) {
  int i = blockIdx.x * blockDim.x + threadIdx.x;
  if (i >= n) return;
  float* rowp = out + (size_t)i * nc;
  float mx = -3.402823466e38f;
  for (int j = 0; j < nc; ++j) mx = fmaxf(mx, rowp[j]);
  float s = 0.f;
  for (int j = 0; j < nc; ++j) s += __expf(rowp[j] - mx);
  float ls = __logf(s);
  for (int j = 0; j < nc; ++j) rowp[j] = rowp[j] - mx - ls;
}

// ---------------------------------------------------------------------------
// Host orchestration (graph-capture safe: only kernel launches on `stream`)
// ---------------------------------------------------------------------------
extern "C" void kernel_launch(void* const* d_in, const int* in_sizes, int n_in,
                              void* d_out, int out_size, void* d_ws, size_t ws_size,
                              hipStream_t stream) {
  const float* x   = (const float*)d_in[0];
  const int*   e1  = (const int*)d_in[1];
  const int*   e2  = (const int*)d_in[2];
  const float* W1  = (const float*)d_in[3];
  const float* b1  = (const float*)d_in[4];
  const float* Wc1 = (const float*)d_in[5];
  const float* bc1 = (const float*)d_in[6];
  const float* Wc2 = (const float*)d_in[7];
  const float* bc2 = (const float*)d_in[8];
  const float* W2  = (const float*)d_in[9];
  const float* b2  = (const float*)d_in[10];

  const int F = 512, H0 = 256, H1 = 128, NC = 40;
  const int N  = in_sizes[0] / F;
  const int E1 = in_sizes[1] / 2;
  const int E2 = in_sizes[2] / 2;
  const int FH = H0 + 2 * H1 + 2 * 128;   // 768 = concat width

  // Workspace carve-up (fp32):
  float* ws      = (float*)d_ws;
  float* final_h = ws;                         // [N, 768] concat buffer
  float* hW      = final_h + (size_t)N * FH;   // [N, 128] shared h@W product
  float* dis1    = hW + (size_t)N * 128;       // [N] deg->rsqrt, edge list 1
  float* dis2    = dis1 + N;                   // [N] deg->rsqrt, edge list 2
  float* out     = (float*)d_out;              // [N, 40]

  const int T = 256;
  auto cdiv = [](long a, long b) { return (int)((a + b - 1) / b); };

  // 1) Degrees and symmetric-norm factors (per edge list, shared by layers).
  zero_kernel<<<cdiv(2L * N, T), T, 0, stream>>>(dis1, 2L * N);
  deg_kernel<<<cdiv(E1, T), T, 0, stream>>>(e1 + E1, E1, dis1);
  deg_kernel<<<cdiv(E2, T), T, 0, stream>>>(e2 + E2, E2, dis2);
  rsqrt_inplace_kernel<<<cdiv(N, T), T, 0, stream>>>(dis1, N);
  rsqrt_inplace_kernel<<<cdiv(N, T), T, 0, stream>>>(dis2, N);

  // 2) h = relu(x @ W_lin1 + b_lin1) -> final_h[:, 0:256]
  {
    dim3 g(cdiv(N, BM), cdiv(H0, BN));
    gemm_bf16_wmma<<<g, T, 0, stream>>>(x, F, 0, W1, H0, b1,
                                        final_h, FH, 0, N, H0, F, 1);
  }

  // 3) Layer 1: hW = h @ W_c1 (computed once, reused by both hops)
  {
    dim3 g(cdiv(N, BM), cdiv(H1, BN));
    gemm_bf16_wmma<<<g, T, 0, stream>>>(final_h, FH, 0, Wc1, H1, nullptr,
                                        hW, H1, 0, N, H1, H0, 0);
  }
  // init R1 slice (cols 256..511) with b_c1, then scatter both edge lists
  set_cols_kernel<<<cdiv((long)N * 256, T), T, 0, stream>>>(
      final_h + H0, FH, bc1, H1, 2 * H1, N);
  gcn_scatter_kernel<<<cdiv(32L * E1, T), T, 0, stream>>>(
      hW, dis1, e1, e1 + E1, E1, final_h + H0, FH);
  gcn_scatter_kernel<<<cdiv(32L * E2, T), T, 0, stream>>>(
      hW, dis2, e2, e2 + E2, E2, final_h + H0 + H1, FH);

  // 4) Layer 2: hW = R1 @ W_c2 (R1 = final_h[:, 256:512])
  {
    dim3 g(cdiv(N, BM), cdiv(H1, BN));
    gemm_bf16_wmma<<<g, T, 0, stream>>>(final_h, FH, H0, Wc2, H1, nullptr,
                                        hW, H1, 0, N, H1, 2 * H1, 0);
  }
  set_cols_kernel<<<cdiv((long)N * 256, T), T, 0, stream>>>(
      final_h + H0 + 2 * H1, FH, bc2, 128, 256, N);
  gcn_scatter_kernel<<<cdiv(32L * E1, T), T, 0, stream>>>(
      hW, dis1, e1, e1 + E1, E1, final_h + H0 + 2 * H1, FH);
  gcn_scatter_kernel<<<cdiv(32L * E2, T), T, 0, stream>>>(
      hW, dis2, e2, e2 + E2, E2, final_h + H0 + 2 * H1 + 128, FH);

  // 5) logits = final_h @ W_lin2 + b_lin2 -> d_out, then in-place log_softmax
  {
    dim3 g(cdiv(N, BM), cdiv(NC, BN));
    gemm_bf16_wmma<<<g, T, 0, stream>>>(final_h, FH, 0, W2, NC, b2,
                                        out, NC, 0, N, NC, FH, 0);
  }
  log_softmax_kernel<<<cdiv(N, T), T, 0, stream>>>(out, N, NC);
}
